// GNNModel_37984690765916
// MI455X (gfx1250) — compile-verified, matrix operand order checked by reading, not run on hardware
//
#include <hip/hip_runtime.h>
#include <hip/hip_bf16.h>

#define N_NODES 100000
#define NEG_SLOPE 0.2f

typedef __attribute__((ext_vector_type(16))) __bf16 v16bf;
typedef __attribute__((ext_vector_type(8)))  __bf16 v8bf;
typedef __attribute__((ext_vector_type(8)))  float  v8f;

// ---------------------------------------------------------------------------
// Convert W [128 x 128] (row-major, [k][n]) to bf16 transposed Wb[n][k]
// so B-fragment loads look exactly like A-fragment loads.
// ---------------------------------------------------------------------------
__global__ void gat_cvtw(const float* __restrict__ W, __bf16* __restrict__ Wb) {
    int t = blockIdx.x * blockDim.x + threadIdx.x;
    if (t >= 128 * 128) return;
    int n = t >> 7, k = t & 127;
    Wb[n * 128 + k] = (__bf16)W[k * 128 + n];
}

// ---------------------------------------------------------------------------
// WMMA GEMM: F[N,128] = H[N,128] @ W[128,128]   (W given as bf16 W^T)
// One block = 16 rows x 128 cols. 8 waves, wave w owns cols [16w,16w+16).
// K loop: 4 x wmma_f32_16x16x32_bf16.
// A-fragment (16-bit A 16x32, wave32): lane<16 -> row=lane, K = kb..kb+7 and
// kb+16..kb+23 with kb = (lane>>4)*8.  B mirrors A with N in place of M.
// ---------------------------------------------------------------------------
__global__ __launch_bounds__(256) void gat_gemm(const float* __restrict__ H,
                                                const __bf16* __restrict__ Wb,
                                                float* __restrict__ F) {
    const int wave = threadIdx.x >> 5;   // 0..7
    const int lane = threadIdx.x & 31;
    const int half = lane >> 4;          // 0 or 1
    const int l16  = lane & 15;
    const int row0 = blockIdx.x * 16;
    const int arow = row0 + l16;
    const int bcol = wave * 16 + l16;
    const int kb   = half * 8;

    v8f c = {};
    #pragma unroll
    for (int k0 = 0; k0 < 128; k0 += 32) {
        // ---- A fragment: f32 -> bf16 conversion in flight
        const float4* ap = reinterpret_cast<const float4*>(H + arow * 128 + k0 + kb);
        float4 a0 = ap[0], a1 = ap[1];   // K = kb .. kb+7
        float4 a2 = ap[4], a3 = ap[5];   // K = kb+16 .. kb+23
        v16bf a;
        a[0]  = (__bf16)a0.x; a[1]  = (__bf16)a0.y; a[2]  = (__bf16)a0.z; a[3]  = (__bf16)a0.w;
        a[4]  = (__bf16)a1.x; a[5]  = (__bf16)a1.y; a[6]  = (__bf16)a1.z; a[7]  = (__bf16)a1.w;
        a[8]  = (__bf16)a2.x; a[9]  = (__bf16)a2.y; a[10] = (__bf16)a2.z; a[11] = (__bf16)a2.w;
        a[12] = (__bf16)a3.x; a[13] = (__bf16)a3.y; a[14] = (__bf16)a3.z; a[15] = (__bf16)a3.w;

        // ---- B fragment: already bf16, 2 x 16B loads
        const __bf16* bp = Wb + bcol * 128 + k0 + kb;
        v8bf blo = *reinterpret_cast<const v8bf*>(bp);
        v8bf bhi = *reinterpret_cast<const v8bf*>(bp + 16);
        v16bf b;
        #pragma unroll
        for (int i = 0; i < 8; ++i) { b[i] = blo[i]; b[8 + i] = bhi[i]; }

        c = __builtin_amdgcn_wmma_f32_16x16x32_bf16(
                /*neg_a=*/false, a, /*neg_b=*/false, b,
                /*c_mod=*/(short)0, c, /*reuse_a=*/false, /*reuse_b=*/false);
    }

    // C/D layout: VGPR v, lane l -> (m = v + 8*half, n = l16)
    float* out = F + (row0 + 8 * half) * 128 + wave * 16 + l16;
    #pragma unroll
    for (int v = 0; v < 8; ++v) out[v * 128] = c[v];
}

// ---------------------------------------------------------------------------
// Per (node, head): el/er dot products; init segment-max key, segment-sum,
// and zero the output accumulator slice.
// ---------------------------------------------------------------------------
__global__ void gat_attn(const float* __restrict__ F,
                         const float* __restrict__ al, const float* __restrict__ ar,
                         float* __restrict__ el, float* __restrict__ er,
                         int* __restrict__ m_int, float* __restrict__ s_sum,
                         float* __restrict__ acc) {
    int t = blockIdx.x * blockDim.x + threadIdx.x;
    if (t >= N_NODES * 4) return;
    int n = t >> 2, h = t & 3;
    const float* fp  = F  + n * 128 + h * 32;
    const float* alp = al + h * 32;
    const float* arp = ar + h * 32;
    float sl = 0.f, sr = 0.f;
    #pragma unroll 8
    for (int d = 0; d < 32; ++d) { float v = fp[d]; sl += v * alp[d]; sr += v * arp[d]; }
    el[t] = sl;
    er[t] = sr;
    m_int[t] = (int)0x80000000;   // INT_MIN == encode(-inf) floor
    s_sum[t] = 0.f;
    float* ap = acc + n * 128 + h * 32;
    #pragma unroll 8
    for (int d = 0; d < 32; ++d) ap[d] = 0.f;
}

__device__ __forceinline__ float leaky(float v) { return v > 0.f ? v : NEG_SLOPE * v; }
// monotone float -> signed-int key (order-preserving), so we can use fast int atomicMax
__device__ __forceinline__ int fkey(float v) {
    int b = __float_as_int(v);
    return b >= 0 ? b : (b ^ 0x7FFFFFFF);
}

// ---------------------------------------------------------------------------
// Edge pass 1: segment max via global_atomic_max_i32 on encoded keys.
// ---------------------------------------------------------------------------
__global__ void gat_edge_max(const int* __restrict__ srcv, const int* __restrict__ dstv,
                             const float* __restrict__ el, const float* __restrict__ er,
                             int* __restrict__ m_int, int E) {
    int t = blockIdx.x * blockDim.x + threadIdx.x;
    if (t >= E * 4) return;
    int e = t >> 2, h = t & 3;
    int s = srcv[e], d = dstv[e];
    float v = leaky(el[s * 4 + h] + er[d * 4 + h]);
    atomicMax(&m_int[d * 4 + h], fkey(v));
}

// Decode keys -> float max (isolated node guard: untouched == INT_MIN -> 0)
__global__ void gat_mdecode(const int* __restrict__ m_int, float* __restrict__ m_f) {
    int t = blockIdx.x * blockDim.x + threadIdx.x;
    if (t >= N_NODES * 4) return;
    int k = m_int[t];
    float m;
    if (k == (int)0x80000000) m = 0.f;
    else {
        int b = k >= 0 ? k : (k ^ 0x7FFFFFFF);
        m = __int_as_float(b);
    }
    m_f[t] = m;
}

// ---------------------------------------------------------------------------
// Edge pass 2: segment sum of exp(e - m[dst]) via global_atomic_add_f32.
// ---------------------------------------------------------------------------
__global__ void gat_edge_sumexp(const int* __restrict__ srcv, const int* __restrict__ dstv,
                                const float* __restrict__ el, const float* __restrict__ er,
                                const float* __restrict__ m_f, float* __restrict__ s_sum, int E) {
    int t = blockIdx.x * blockDim.x + threadIdx.x;
    if (t >= E * 4) return;
    int e = t >> 2, h = t & 3;
    int s = srcv[e], d = dstv[e];
    float v = leaky(el[s * 4 + h] + er[d * 4 + h]);
    atomicAdd(&s_sum[d * 4 + h], __expf(v - m_f[d * 4 + h]));
}

// ---------------------------------------------------------------------------
// Edge pass 3: one wave32 per edge; lane owns 4 consecutive features (one
// head per lane). alpha = exp(e-m)/(s+eps); acc[dst] += alpha * f[src].
// f and acc are L2-resident (51 MB each, 192 MB L2).
// ---------------------------------------------------------------------------
__global__ __launch_bounds__(256) void gat_edge_msg(
        const int* __restrict__ srcv, const int* __restrict__ dstv,
        const float* __restrict__ F,
        const float* __restrict__ el, const float* __restrict__ er,
        const float* __restrict__ m_f, const float* __restrict__ s_sum,
        float* __restrict__ acc, int E) {
    int wid = (blockIdx.x * blockDim.x + threadIdx.x) >> 5;
    if (wid >= E) return;
    int lane = threadIdx.x & 31;
    int s = srcv[wid], d = dstv[wid];
    int h = lane >> 3;                       // (lane*4)/32
    float v = leaky(el[s * 4 + h] + er[d * 4 + h]);
    float alpha = __expf(v - m_f[d * 4 + h]) / (s_sum[d * 4 + h] + 1e-9f);
    const float4 fv = *reinterpret_cast<const float4*>(F + s * 128 + lane * 4);
    float* ap = acc + d * 128 + lane * 4;
    atomicAdd(ap + 0, fv.x * alpha);
    atomicAdd(ap + 1, fv.y * alpha);
    atomicAdd(ap + 2, fv.z * alpha);
    atomicAdd(ap + 3, fv.w * alpha);
}

__global__ void gat_relu(const float* __restrict__ in, float* __restrict__ out, int n) {
    int t = blockIdx.x * blockDim.x + threadIdx.x;
    if (t < n) out[t] = fmaxf(in[t], 0.f);
}

// ---------------------------------------------------------------------------
extern "C" void kernel_launch(void* const* d_in, const int* in_sizes, int n_in,
                              void* d_out, int out_size, void* d_ws, size_t ws_size,
                              hipStream_t stream) {
    const float* x  = (const float*)d_in[0];
    const int*   ei = (const int*)d_in[1];
    const int E = in_sizes[1] / 2;
    const int* srcv = ei;
    const int* dstv = ei + E;

    const float* W[3]  = { (const float*)d_in[2], (const float*)d_in[5], (const float*)d_in[8]  };
    const float* al[3] = { (const float*)d_in[3], (const float*)d_in[6], (const float*)d_in[9]  };
    const float* ar[3] = { (const float*)d_in[4], (const float*)d_in[7], (const float*)d_in[10] };

    // ---- workspace carve-up
    char* p = (char*)d_ws;
    const size_t featB = (size_t)N_NODES * 128 * sizeof(float);   // 51.2 MB
    const size_t nhB   = (size_t)N_NODES * 4 * sizeof(float);     // 1.6 MB
    float*  f     = (float*)p;  p += featB;
    float*  big1  = (float*)p;  p += featB;
    float*  big2  = (float*)p;  p += featB;
    float*  el    = (float*)p;  p += nhB;
    float*  er    = (float*)p;  p += nhB;
    int*    m_int = (int*)p;    p += nhB;
    float*  m_f   = (float*)p;  p += nhB;
    float*  s_sum = (float*)p;  p += nhB;
    __bf16* Wb    = (__bf16*)p; p += 128 * 128 * sizeof(__bf16);

    const int TB = 256;
    const int gW    = (128 * 128 + TB - 1) / TB;
    const int gGemm = N_NODES / 16;                 // 6250, exact
    const int gNH   = (N_NODES * 4 + TB - 1) / TB;
    const int gE4   = (E * 4 + TB - 1) / TB;
    const int gMsg  = (E * 32 + TB - 1) / TB;       // wave per edge
    const int gFeat = (N_NODES * 128 + TB - 1) / TB;

    const float* h = x;                   // layer input
    float* accs[3] = { big1, big2, big1 };

    for (int l = 0; l < 3; ++l) {
        float* acc = accs[l];
        gat_cvtw      <<<gW,    TB, 0, stream>>>(W[l], Wb);
        gat_gemm      <<<gGemm, TB, 0, stream>>>(h, Wb, f);
        gat_attn      <<<gNH,   TB, 0, stream>>>(f, al[l], ar[l], el, er, m_int, s_sum, acc);
        gat_edge_max  <<<gE4,   TB, 0, stream>>>(srcv, dstv, el, er, m_int, E);
        gat_mdecode   <<<gNH,   TB, 0, stream>>>(m_int, m_f);
        gat_edge_sumexp<<<gE4,  TB, 0, stream>>>(srcv, dstv, el, er, m_f, s_sum, E);
        gat_edge_msg  <<<gMsg,  TB, 0, stream>>>(srcv, dstv, f, el, er, m_f, s_sum, acc, E);
        float* outp = (l == 2) ? (float*)d_out : acc;   // relu (in place for l<2)
        gat_relu      <<<gFeat, TB, 0, stream>>>(acc, outp, N_NODES * 128);
        h = acc;
    }
}
